// BasalGangliaActor_48052094108013
// MI455X (gfx1250) — compile-verified, matrix operand order checked by reading, not run on hardware
//
#include <hip/hip_runtime.h>

// ---------------------------------------------------------------------------
// Basal-ganglia LIF actor for MI455X (gfx1250, wave32).
//
// The T=2000 recurrence is sequential -> all neuron state lives in VGPRs for
// the whole simulation; one wave simulates 16 batch elements. The two
// [16x16]x[16x4] spike projections per step run on V_WMMA_F32_16X16X32_F16
// (K padded 16->32, N padded 4->16 with zeros); spikes are exact 0/1 in f16.
// Forward spike == heaviside(v - 1) (surrogate terms cancel in value), and
// float32(1/1.00000000001) == 1.0f, so LIF is sub/add/cmp/select.
//
// Latency plan: only ~1 wave/SIMD is resident, so the serial loop body must
// hide its own memory latency. Input is register double-buffered (loads for
// t+1 issue before the ~380-op compute of t) and global_prefetch_b8 runs 8
// timesteps (8 KB/wave) ahead into L2/WGP$.
// ---------------------------------------------------------------------------

typedef __attribute__((ext_vector_type(16))) _Float16 v16h;
typedef __attribute__((ext_vector_type(8)))  float    v8f;

#define LIF_INV_TAU 1.0f   // float32(1 / 1.00000000001) rounds to exactly 1.0f
#define LIF_V_TH    1.0f
#define PF_DIST     8      // prefetch distance in timesteps

// Single-step LIF (decay_input=True, hard reset to 0). Returns spike (0/1).
__device__ __forceinline__ float lif_step(float& v, float x) {
    v = v + (x - v) * LIF_INV_TAU;
    const bool fired = (v >= LIF_V_TH);     // heaviside(v - v_th, 1.0)
    const float s = fired ? 1.0f : 0.0f;
    v = fired ? 0.0f : v;                   // hard reset, detach_reset
    return s;
}

__global__ __launch_bounds__(32)
void bg_actor_kernel(const float* __restrict__ pfc,   // [T, B, 16]
                     const float* __restrict__ w1,    // [16, 4]
                     const float* __restrict__ w2,    // [16, 4]
                     float* __restrict__ out,         // [B] action + [B,4] totals
                     int T, int B) {
    const int lane       = threadIdx.x;        // 0..31
    const int half       = lane >> 4;          // 0: states/K 0-7, 1: states/K 8-15
    const int sub        = lane & 15;          // A: batch row m   B/D: column n
    const int batch_base = blockIdx.x * 16;

    // ---- per-lane fixed pathway weights (action = D-matrix column sub) ----
    const int   a   = sub & 3;                 // clamp; lanes with sub>=4 idle
    const float w1b = (a == 0) ? 0.2f  : (a == 1) ? 0.12f : (a == 2) ? 0.07f : 0.03f;
    const float w2b = (a == 0) ? 0.03f : (a == 1) ? 0.07f : (a == 2) ? 0.12f : 0.2f;
    const float W_D1_GPI   = -5.0f * w1b;
    const float W_D2_GPE   = -5.0f * w1b;
    const float W_GPE_STN  = -4.0f * w1b;
    const float W_STN_GPI  =  6.0f * w1b;
    const float W_GPI_THAL = -5.0f * w2b;
    const float W_THAL_PMC =  5.0f * w1b;

    // ---- build f16 B-matrices (32x16, column n in lane n / n+16) ----------
    // halves 0..7 = K(8*half)..K(8*half+7); halves 8..15 = K 16..31 pad (zero)
    v16h bm1 = {};
    v16h bm2 = {};
    if (sub < 4) {
        #pragma unroll
        for (int k = 0; k < 8; ++k) {
            const int kk = half * 8 + k;                 // state index
            bm1[k] = (_Float16)w1[kk * 4 + sub];
            bm2[k] = (_Float16)w2[kk * 4 + sub];
        }
    }

    // ---- all recurrent state in VGPRs -------------------------------------
    float v_pfc[8]  = {};                                // 8 PFC neurons/lane
    float v_d1[8]   = {}, v_d2[8]  = {}, v_gpe[8] = {};
    float v_stn[8]  = {}, v_gpi[8] = {}, v_thal[8] = {};
    float v_pmc[8]  = {}, tot[8]   = {};
    v16h  av = {};                                       // A: halves 8..15 stay 0

    const size_t stride = (size_t)B * 16;                // floats per timestep
    const size_t row    = ((size_t)(batch_base + sub)) * 16 + (size_t)half * 8;
    const float* p      = pfc + row;

    // prologue of the register double-buffer: loads for t = 0
    float4 x0 = *(const float4*)(p);
    float4 x1 = *(const float4*)(p + 4);

    for (int t = 0; t < T; ++t) {
        // deep prefetch into L2/WGP$ (global_prefetch_b8), 8 steps ahead
        if (t + PF_DIST < T) __builtin_prefetch(p + (size_t)PF_DIST * stride, 0, 0);

        // issue next-step loads *before* the long serial compute (branch-free:
        // last iteration just re-reads the current row)
        const float* pn = p + ((t + 1 < T) ? stride : 0);
        const float4 n0 = *(const float4*)(pn);
        const float4 n1 = *(const float4*)(pn + 4);

        const float xs[8] = {x0.x, x0.y, x0.z, x0.w, x1.x, x1.y, x1.z, x1.w};

        // PFC LIF -> spike bits directly into WMMA A layout
        #pragma unroll
        for (int i = 0; i < 8; ++i) {
            av[i] = (_Float16)lif_step(v_pfc[i], xs[i]);
        }

        // d1_c / d2_c = spikes @ W  (16x16 result, N 0..3 valid)
        const v8f zero = {};
        v8f c1 = __builtin_amdgcn_wmma_f32_16x16x32_f16(
            false, av, false, bm1, (short)0, zero, false, false);
        v8f c2 = __builtin_amdgcn_wmma_f32_16x16x32_f16(
            false, av, false, bm2, (short)0, zero, false, false);

        // downstream 7-layer chain, run in-place in the D layout:
        // component r of this lane = batch (8*half + r), action sub
        #pragma unroll
        for (int r = 0; r < 8; ++r) {
            const float s_d1   = lif_step(v_d1[r],  c1[r]);
            const float s_d2   = lif_step(v_d2[r],  c2[r]);
            const float direct = s_d1 * W_D1_GPI;
            const float s_gpe  = lif_step(v_gpe[r],  s_d2  * W_D2_GPE   + 1.5f);
            const float s_stn  = lif_step(v_stn[r],  s_gpe * W_GPE_STN  + 1.8f);
            const float s_gpi  = lif_step(v_gpi[r],  direct + s_stn * W_STN_GPI + 1.5f);
            const float s_thal = lif_step(v_thal[r], s_gpi * W_GPI_THAL + 1.2f);
            const float s_pmc  = lif_step(v_pmc[r],  s_thal * W_THAL_PMC + 0.5f);
            tot[r] += s_pmc;
        }

        // rotate the double buffer
        x0 = n0;
        x1 = n1;
        p  = pn;
    }

    // ---- outputs: [B] argmax action, then [B,4] spike totals ---------------
    __shared__ float sp[16][4];
    if (sub < 4) {
        #pragma unroll
        for (int r = 0; r < 8; ++r) {
            const int mm = half * 8 + r;                 // batch within wave
            const int b  = batch_base + mm;
            out[(size_t)B + (size_t)b * 4 + sub] = tot[r];
            sp[mm][sub] = tot[r];
        }
    }
    __syncthreads();
    if (lane < 16) {
        const int b = batch_base + lane;
        float best = sp[lane][0];
        int   bi   = 0;
        #pragma unroll
        for (int j = 1; j < 4; ++j) {
            const float v = sp[lane][j];
            if (v > best) { best = v; bi = j; }          // first-max like argmax
        }
        out[b] = (float)bi;
    }
}

extern "C" void kernel_launch(void* const* d_in, const int* in_sizes, int n_in,
                              void* d_out, int out_size, void* d_ws, size_t ws_size,
                              hipStream_t stream) {
    const float* pfc = (const float*)d_in[0];   // [T, B, 16]
    const float* w1  = (const float*)d_in[1];   // [16, 4]
    const float* w2  = (const float*)d_in[2];   // [16, 4]
    float* out = (float*)d_out;                 // [B] + [B,4] concatenated

    const int B = out_size / 5;                 // action + 4 totals per batch
    const int T = in_sizes[0] / (B * 16);
    const int grid = B / 16;                    // one wave per 16 batch rows

    bg_actor_kernel<<<grid, 32, 0, stream>>>(pfc, w1, w2, out, T, B);
}
